// Attention_34119220200087
// MI455X (gfx1250) — compile-verified, matrix operand order checked by reading, not run on hardware
//
#include <hip/hip_runtime.h>
#include <cstdint>

// CDNA5 / gfx1250: wave32, WMMA 16x16x32 bf16 -> f32
typedef __attribute__((ext_vector_type(16))) __bf16 bf16x16;
typedef __attribute__((ext_vector_type(8)))  float  f32x8;

#define WMMA_BF16(a, b, c) \
  __builtin_amdgcn_wmma_f32_16x16x32_bf16(false, (a), false, (b), (short)0, (c), false, false)

static __device__ __forceinline__ __bf16 to_bf16(float f) { return (__bf16)f; }

// ---------------------------------------------------------------------------
// Kernel 0: pre-pack weights into B-fragment-ordered bf16.
//   wpk : [mat(4)][kc(4)][ct(16)][lane(32)][e(16)]  from wq/wk/wv/wg [128,256]
//   wopk: [kc(8)][ct(8)][lane(32)][e(16)]           from wo [256,128]
// B-frag element e of lane l for chunk kc, coltile ct:
//   n = l&15, half = l>>4, kk = kc*32 + ((e>>3)<<4) + half*8 + (e&7)
//   value = W[kk][ct*16 + n]
// ---------------------------------------------------------------------------
__global__ void __launch_bounds__(256) pack_w_kernel(
    const float* __restrict__ wq, const float* __restrict__ wk,
    const float* __restrict__ wv, const float* __restrict__ wg,
    const float* __restrict__ wo,
    __bf16* __restrict__ wpk, __bf16* __restrict__ wopk)
{
  const int idx = blockIdx.x * 256 + threadIdx.x;   // 640 blocks -> 163840
  if (idx < 131072) {
    const int e    = idx & 15;
    const int lane = (idx >> 4) & 31;
    const int ct   = (idx >> 9) & 15;
    const int kc   = (idx >> 13) & 3;
    const int mat  = idx >> 15;
    const int n    = lane & 15, half = lane >> 4;
    const int kk   = kc * 32 + ((e >> 3) << 4) + half * 8 + (e & 7);
    const float* w = (mat == 0) ? wq : (mat == 1) ? wk : (mat == 2) ? wv : wg;
    wpk[idx] = to_bf16(w[(size_t)kk * 256 + ct * 16 + n]);
  } else {
    const int i    = idx - 131072;
    const int e    = i & 15;
    const int lane = (i >> 4) & 31;
    const int ct   = (i >> 9) & 7;
    const int kc   = i >> 12;
    const int n    = lane & 15, half = lane >> 4;
    const int kk   = kc * 32 + ((e >> 3) << 4) + half * 8 + (e & 7);
    wopk[i] = to_bf16(wo[(size_t)kk * 128 + ct * 16 + n]);
  }
}

// ---------------------------------------------------------------------------
// Kernel 1: QKV + gate projections.
//   x[65536,128] @ W[128,256]  (4 matrices: wq,wk,wv,wg)
// One wave per 16x16 output tile, K-loop of 4 chunks of 32.
// B-fragments come pre-packed (one 32B load/lane/WMMA).
// q,k -> bf16 row-major [n,h,s,32] ; v -> bf16 transposed [n,h,32,s] ;
// gate -> f32 [row, 256] = sigmoid(x@wg + bg)
// ---------------------------------------------------------------------------
__global__ void __launch_bounds__(256) proj_kernel(
    const float* __restrict__ xq, const float* __restrict__ xk,
    const float* __restrict__ xv,
    const __bf16* __restrict__ wpk, const float* __restrict__ bg,
    __bf16* __restrict__ qb, __bf16* __restrict__ kb,
    __bf16* __restrict__ vt, float* __restrict__ gate)
{
  const int wid  = threadIdx.x >> 5;
  const int lane = threadIdx.x & 31;
  const int w    = blockIdx.x * 8 + wid;      // 2^18 waves total
  const int ct   = w & 15;                    // column tile (of 16)
  const int rt   = (w >> 4) & 4095;           // row tile (of 4096)
  const int mat  = w >> 16;                   // 0=q 1=k 2=v 3=gate
  const int nl   = lane & 15;
  const int half = lane >> 4;
  const int sbase = rt * 16;

  const float* x = (mat == 1) ? xk : (mat == 2) ? xv : xq;

  f32x8 acc = {};
  const float* arow = x + (size_t)(sbase + nl) * 128;
  const int nn = ct * 16 + nl;                // output column 0..255

  for (int kc = 0; kc < 4; ++kc) {
    bf16x16 a;
    const int k0 = kc * 32 + half * 8;
#pragma unroll
    for (int e = 0; e < 8; ++e) {
      a[e]     = to_bf16(arow[k0 + e]);
      a[8 + e] = to_bf16(arow[k0 + 16 + e]);
    }
    const bf16x16 b = *(const bf16x16*)(wpk +
        ((size_t)(((mat * 4 + kc) * 16 + ct) * 32 + lane) << 4));
    acc = WMMA_BF16(a, b, acc);
  }

  // D layout: element r of this lane = row (r + 8*half), col nn
  const int row0 = sbase + half * 8;          // global flat row for r=0
  const int nidx = row0 >> 8;                 // which of the 256 problems
  const int s0   = row0 & 255;
  const int hh   = nn >> 5;                   // head
  const int cc   = nn & 31;                   // channel within head

  if (mat == 0) {
    const float sc = 0.17677669529663687f;    // 1/sqrt(32)
    const size_t base = ((size_t)(nidx * 8 + hh) * 256 + s0) * 32 + cc;
#pragma unroll
    for (int r = 0; r < 8; ++r) qb[base + (size_t)r * 32] = to_bf16(acc[r] * sc);
  } else if (mat == 1) {
    const size_t base = ((size_t)(nidx * 8 + hh) * 256 + s0) * 32 + cc;
#pragma unroll
    for (int r = 0; r < 8; ++r) kb[base + (size_t)r * 32] = to_bf16(acc[r]);
  } else if (mat == 2) {
    const size_t base = ((size_t)(nidx * 8 + hh) * 32 + cc) * 256 + s0;
#pragma unroll
    for (int r = 0; r < 8; ++r) vt[base + r] = to_bf16(acc[r]);  // contiguous
  } else {
    const size_t base = (size_t)row0 * 256 + nn;
    const float bgv = bg[nn];
#pragma unroll
    for (int r = 0; r < 8; ++r)
      gate[base + (size_t)r * 256] = 1.0f / (1.0f + __expf(-(acc[r] + bgv)));
  }
}

// ---------------------------------------------------------------------------
// Kernel 2: attention per (n, h, 16-row q-tile).
// logits (16x256) fully in registers (16 WMMAs, C=32 = one K-step),
// + bias + (mask-1)*1e9, softmax (shfl within 16-lane halves),
// normalized probs -> LDS bf16 row-major, then P@V via 16 WMMAs,
// gated, bf16 store to og[row,256].
// ---------------------------------------------------------------------------
__global__ void __launch_bounds__(128) attn_kernel(
    const float* __restrict__ mask, const float* __restrict__ bias,
    const __bf16* __restrict__ qb, const __bf16* __restrict__ kb,
    const __bf16* __restrict__ vt, const float* __restrict__ gate,
    __bf16* __restrict__ og)
{
  __shared__ __bf16 sp[4][16][256];           // 32 KB: per-wave prob tile
  const int wid  = threadIdx.x >> 5;
  const int lane = threadIdx.x & 31;
  const int w    = blockIdx.x * 4 + wid;      // 32768 waves
  const int qt   = w & 15;
  const int h    = (w >> 4) & 7;
  const int n    = w >> 7;
  const int nl   = lane & 15;
  const int half = lane >> 4;

  const __bf16* qhp = qb + (size_t)(n * 8 + h) * 256 * 32;
  const __bf16* khp = kb + (size_t)(n * 8 + h) * 256 * 32;
  const __bf16* vtp = vt + (size_t)(n * 8 + h) * 32 * 256;
  const float*  mp  = mask + (size_t)n * 256;
  const float*  bp  = bias + (size_t)h * 256 * 256;

  // A fragment: q rows qt*16..+15
  bf16x16 aq;
  {
    const __bf16* qr = qhp + (size_t)(qt * 16 + nl) * 32 + half * 8;
#pragma unroll
    for (int e = 0; e < 8; ++e) { aq[e] = qr[e]; aq[8 + e] = qr[16 + e]; }
  }

  // Logits: 16 tiles of 16x16
  f32x8 lg[16];
#pragma unroll
  for (int t = 0; t < 16; ++t) {
    bf16x16 bk;
    const __bf16* kr = khp + (size_t)(t * 16 + nl) * 32 + half * 8;
#pragma unroll
    for (int e = 0; e < 8; ++e) { bk[e] = kr[e]; bk[8 + e] = kr[16 + e]; }
    f32x8 c = {};
    lg[t] = WMMA_BF16(aq, bk, c);
  }

  // bias + mask, then row softmax (row m = r + 8*half, col = t*16 + nl)
  float rmax[8];
#pragma unroll
  for (int r = 0; r < 8; ++r) rmax[r] = -3.0e38f;
#pragma unroll
  for (int t = 0; t < 16; ++t) {
    const int kc = t * 16 + nl;
    const float mk = (mp[kc] - 1.0f) * 1.0e9f;
#pragma unroll
    for (int r = 0; r < 8; ++r) {
      const int qg = qt * 16 + r + 8 * half;
      float v = lg[t][r] + mk + bp[(size_t)qg * 256 + kc];
      lg[t][r] = v;
      rmax[r] = fmaxf(rmax[r], v);
    }
  }
#pragma unroll
  for (int r = 0; r < 8; ++r) {
    float v = rmax[r];
    for (int m = 1; m < 16; m <<= 1) v = fmaxf(v, __shfl_xor(v, m, 32));
    rmax[r] = v;
  }
  float rsum[8];
#pragma unroll
  for (int r = 0; r < 8; ++r) rsum[r] = 0.0f;
#pragma unroll
  for (int t = 0; t < 16; ++t)
#pragma unroll
    for (int r = 0; r < 8; ++r) {
      float e = __expf(lg[t][r] - rmax[r]);
      lg[t][r] = e;
      rsum[r] += e;
    }
#pragma unroll
  for (int r = 0; r < 8; ++r) {
    float v = rsum[r];
    for (int m = 1; m < 16; m <<= 1) v += __shfl_xor(v, m, 32);
    rsum[r] = 1.0f / v;
  }
#pragma unroll
  for (int t = 0; t < 16; ++t)
#pragma unroll
    for (int r = 0; r < 8; ++r)
      sp[wid][r + 8 * half][t * 16 + nl] = to_bf16(lg[t][r] * rsum[r]);

  __syncthreads();

  // o = P @ V : 8 K-chunks of 32, 2 column tiles (C=32)
  f32x8 o0 = {}, o1 = {};
#pragma unroll
  for (int kc = 0; kc < 8; ++kc) {
    bf16x16 ap, bv;
    const __bf16* pr = &sp[wid][nl][kc * 32 + half * 8];
#pragma unroll
    for (int e = 0; e < 8; ++e) { ap[e] = pr[e]; ap[8 + e] = pr[16 + e]; }
    const __bf16* vr0 = vtp + (size_t)nl * 256 + kc * 32 + half * 8;
#pragma unroll
    for (int e = 0; e < 8; ++e) { bv[e] = vr0[e]; bv[8 + e] = vr0[16 + e]; }
    o0 = WMMA_BF16(ap, bv, o0);
    const __bf16* vr1 = vtp + (size_t)(16 + nl) * 256 + kc * 32 + half * 8;
#pragma unroll
    for (int e = 0; e < 8; ++e) { bv[e] = vr1[e]; bv[8 + e] = vr1[16 + e]; }
    o1 = WMMA_BF16(ap, bv, o1);
  }

  // gate + store (og row-major [n*256+s][256])
#pragma unroll
  for (int r = 0; r < 8; ++r) {
    const int sg = qt * 16 + r + 8 * half;
    const size_t rowb = ((size_t)n * 256 + sg) * 256;
    const int d0 = h * 32 + nl;
    const float g0 = gate[rowb + d0];
    og[rowb + d0] = to_bf16(o0[r] * g0);
    const int d1 = h * 32 + 16 + nl;
    const float g1 = gate[rowb + d1];
    og[rowb + d1] = to_bf16(o1[r] * g1);
  }
}

// ---------------------------------------------------------------------------
// Kernel 3: out = og[65536,256] @ wo[256,128] + bo  (wo pre-packed bf16)
// ---------------------------------------------------------------------------
__global__ void __launch_bounds__(256) oproj_kernel(
    const __bf16* __restrict__ og, const __bf16* __restrict__ wopk,
    const float* __restrict__ bo, float* __restrict__ out)
{
  const int wid  = threadIdx.x >> 5;
  const int lane = threadIdx.x & 31;
  const int w    = blockIdx.x * 8 + wid;      // 32768 waves
  const int ct   = w & 7;
  const int rt   = w >> 3;
  const int nl   = lane & 15;
  const int half = lane >> 4;

  f32x8 acc = {};
  const __bf16* arow = og + (size_t)(rt * 16 + nl) * 256;
  const int nn = ct * 16 + nl;
  for (int kc = 0; kc < 8; ++kc) {
    bf16x16 a;
    const int k0 = kc * 32 + half * 8;
#pragma unroll
    for (int e = 0; e < 8; ++e) { a[e] = arow[k0 + e]; a[8 + e] = arow[k0 + 16 + e]; }
    const bf16x16 b = *(const bf16x16*)(wopk +
        ((size_t)((kc * 8 + ct) * 32 + lane) << 4));
    acc = WMMA_BF16(a, b, acc);
  }
  const int row0 = rt * 16 + half * 8;
  const float bb = bo[nn];
#pragma unroll
  for (int r = 0; r < 8; ++r) out[(size_t)(row0 + r) * 128 + nn] = acc[r] + bb;
}

// ---------------------------------------------------------------------------
// Workspace layout (bytes):
//   qb   bf16 [256*8*256*32]                 0 .. 33554432
//   kb   bf16                         33554432 .. 67108864
//   vt   bf16 (V transposed)          67108864 .. 100663296
//   og   bf16 [65536*256]            100663296 .. 134217728
//   gate f32  [65536*256]            134217728 .. 201326592
//   wpk  bf16 [131072]               201326592 .. 201588736
//   wopk bf16 [32768]                201588736 .. 201654272
// ---------------------------------------------------------------------------
extern "C" void kernel_launch(void* const* d_in, const int* in_sizes, int n_in,
                              void* d_out, int out_size, void* d_ws, size_t ws_size,
                              hipStream_t stream) {
  const float* xq   = (const float*)d_in[0];
  const float* xk   = (const float*)d_in[1];
  const float* xv   = (const float*)d_in[2];
  const float* mask = (const float*)d_in[3];
  const float* bias = (const float*)d_in[4];
  const float* wq   = (const float*)d_in[5];
  const float* wk   = (const float*)d_in[6];
  const float* wv   = (const float*)d_in[7];
  const float* wg   = (const float*)d_in[8];
  const float* bg   = (const float*)d_in[9];
  const float* wo   = (const float*)d_in[10];
  const float* bo   = (const float*)d_in[11];

  char* ws = (char*)d_ws;
  __bf16* qb   = (__bf16*)(ws);
  __bf16* kb   = (__bf16*)(ws + 33554432);
  __bf16* vt   = (__bf16*)(ws + 67108864);
  __bf16* og   = (__bf16*)(ws + 100663296);
  float*  gate = (float*)(ws + 134217728);
  __bf16* wpk  = (__bf16*)(ws + 201326592);
  __bf16* wopk = (__bf16*)(ws + 201588736);

  pack_w_kernel<<<640, 256, 0, stream>>>(wq, wk, wv, wg, wo, wpk, wopk);
  proj_kernel<<<32768, 256, 0, stream>>>(xq, xk, xv, wpk, bg, qb, kb, vt, gate);
  attn_kernel<<<8192, 128, 0, stream>>>(mask, bias, qb, kb, vt, gate, og);
  oproj_kernel<<<4096, 256, 0, stream>>>(og, wopk, bo, (float*)d_out);
}
